// WindowAttention_6554120094263
// MI455X (gfx1250) — compile-verified, matrix operand order checked by reading, not run on hardware
//
#include <hip/hip_runtime.h>

typedef _Float16 v16h __attribute__((ext_vector_type(16)));
typedef _Float16 v8h  __attribute__((ext_vector_type(8)));
typedef float    v8f  __attribute__((ext_vector_type(8)));

#define XS_STRIDE 520   // 512 + 8 halves pad (16B aligned rows, bank-spread)
#define HS_STRIDE 72    // 64 + 8
#define SF_STRIDE 68

union V16U { v16h v; v8h h[2]; };

// Load one 16x32 f16 operand (A layout; B^T uses the identical pattern).
// p0 points at element [row0][k0] of the sub-block; rows are `stride` halves apart.
__device__ __forceinline__ v16h frag_ld(const _Float16* p0, int stride) {
  const int lane = threadIdx.x & 31;
  const _Float16* p = p0 + (lane & 15) * stride + ((lane >> 4) << 3);
  V16U u;
  u.h[0] = *(const v8h*)(p);        // K = kb .. kb+7
  u.h[1] = *(const v8h*)(p + 16);   // K = kb+16 .. kb+23
  return u.v;
}

__device__ __forceinline__ v8f wmma_f16(v16h a, v16h b, v8f c) {
  return __builtin_amdgcn_wmma_f32_16x16x32_f16(false, a, false, b, (short)0, c,
                                                false, false);
}

__global__ void convert_weights(const float* __restrict__ wqkv,
                                const float* __restrict__ outw,
                                _Float16* __restrict__ wq_h,
                                _Float16* __restrict__ ow_h) {
  const int i = blockIdx.x * blockDim.x + threadIdx.x;
  if (i < 3 * 512 * 512) wq_h[i] = (_Float16)wqkv[i];
  if (i < 512 * 512)     ow_h[i] = (_Float16)outw[i];
}

// One 64x64 output (16 tiles) of X @ W^T for matrix MAT (0=Q,1=K,2=V), head h.
// Wave blocking: tm = wave>>1, two n-tiles per wave sharing the A fragment.
// MAT compile-time -> straight-line stores (no EXEC divergence).
template<int MAT>
__device__ __forceinline__ void qkv_gemm_head(
    const _Float16* Xs, const _Float16* __restrict__ wqkv_h,
    const float* __restrict__ wqkv_b, int h,
    _Float16* Qh, _Float16* Kh, _Float16* Vt,
    int wave, int nl, int half) {
  const int tm  = wave >> 1;
  const int tn0 = (wave & 1) * 2;
  v8f acc0 = {}, acc1 = {};
  const _Float16* ab  = Xs + tm * 16 * XS_STRIDE;
  const _Float16* wb0 = wqkv_h + (size_t)(MAT * 512 + h * 64 + tn0 * 16) * 512;
  const _Float16* wb1 = wb0 + 16 * 512;
  #pragma unroll 4
  for (int ks = 0; ks < 16; ++ks) {
    v16h a  = frag_ld(ab  + ks * 32, XS_STRIDE);
    v16h b0 = frag_ld(wb0 + ks * 32, 512);
    v16h b1 = frag_ld(wb1 + ks * 32, 512);
    acc0 = wmma_f16(a, b0, acc0);
    acc1 = wmma_f16(a, b1, acc1);
  }
  #pragma unroll
  for (int u = 0; u < 2; ++u) {
    const v8f acc = u ? acc1 : acc0;
    const int tn = tn0 + u;
    const int n  = tn * 16 + nl;
    const float bias = wqkv_b[MAT * 512 + h * 64 + n];
    if (MAT == 2) {
      // V stored transposed: consecutive m (=token) -> consecutive Vt columns
      v8h hv;
      #pragma unroll
      for (int r = 0; r < 8; ++r) hv[r] = (_Float16)(acc[r] + bias);
      *(v8h*)(Vt + n * HS_STRIDE + tm * 16 + 8 * half) = hv;
    } else {
      const float scale = (MAT == 1) ? 0.125f : 1.0f;  // fold 1/sqrt(64) into K
      _Float16* dst = (MAT == 0) ? Qh : Kh;
      #pragma unroll
      for (int r = 0; r < 8; ++r)
        dst[(tm * 16 + r + 8 * half) * HS_STRIDE + n] =
            (_Float16)((acc[r] + bias) * scale);
    }
  }
}

__global__ __launch_bounds__(256)
void swin_attn_kernel(const float* __restrict__ x,
                      const float* __restrict__ mask,
                      const _Float16* __restrict__ wqkv_h,
                      const float* __restrict__ wqkv_b,
                      const float* __restrict__ rel_table,
                      const _Float16* __restrict__ outw_h,
                      const float* __restrict__ out_b,
                      float* __restrict__ out) {
  __shared__ _Float16 Xs[64 * XS_STRIDE];  // gathered window, f16
  __shared__ _Float16 Ys[64 * XS_STRIDE];  // attention output (all heads), f16
  __shared__ _Float16 Qh[64 * HS_STRIDE];  // Q[head]  row-major (token, d)
  __shared__ _Float16 Kh[64 * HS_STRIDE];  // K[head]/8 row-major (token, d)
  __shared__ _Float16 Vt[64 * HS_STRIDE];  // V[head]^T row-major (d, token)
  __shared__ _Float16 Sh[64 * HS_STRIDE];  // softmaxed scores, f16
  __shared__ float    Sf[64 * SF_STRIDE];  // scores + bias + mask, f32

  const int tid  = threadIdx.x;
  const int wave = tid >> 5;
  const int lane = tid & 31;
  const int nl   = lane & 15;
  const int half = lane >> 4;

  const int b  = blockIdx.x >> 6;
  const int w  = blockIdx.x & 63;
  const int wr = w >> 3, wc = w & 7;

  // ---- roll(-4,-4) + window partition gather, f32 -> f16 into LDS
  {
    const int tok = tid >> 2;            // token 0..63
    const int sub = tid & 3;             // 128-channel quarter
    const int tr = tok >> 3, tc = tok & 7;
    const int sr = (wr * 8 + tr + 4) & 63;
    const int sc = (wc * 8 + tc + 4) & 63;
    const float* xp = x + (((size_t)b * 4096) + sr * 64 + sc) * 512 + sub * 128;
    _Float16* xd = Xs + tok * XS_STRIDE + sub * 128;
    #pragma unroll 8
    for (int c = 0; c < 128; c += 4) {
      float4 v = *(const float4*)(xp + c);
      xd[c + 0] = (_Float16)v.x; xd[c + 1] = (_Float16)v.y;
      xd[c + 2] = (_Float16)v.z; xd[c + 3] = (_Float16)v.w;
    }
  }
  __syncthreads();

  const float* maskw = mask + (size_t)w * 4096;

  for (int h = 0; h < 8; ++h) {
    // ---- per-head Q, K/8, V^T (each wave: 2 tiles per matrix, shared A frag)
    qkv_gemm_head<0>(Xs, wqkv_h, wqkv_b, h, Qh, Kh, Vt, wave, nl, half);
    qkv_gemm_head<1>(Xs, wqkv_h, wqkv_b, h, Qh, Kh, Vt, wave, nl, half);
    qkv_gemm_head<2>(Xs, wqkv_h, wqkv_b, h, Qh, Kh, Vt, wave, nl, half);
    __syncthreads();

    // ---- S[i][j] = (K/8)_i . Q_j + relbias[i,j] + mask[i,j]  (quirk preserved)
    {
      const int tm = wave >> 1, tn0 = (wave & 1) * 2;
      v8f acc0 = {}, acc1 = {};
      #pragma unroll
      for (int ks = 0; ks < 2; ++ks) {
        v16h a  = frag_ld(Kh + tm * 16 * HS_STRIDE + ks * 32, HS_STRIDE);
        v16h b0 = frag_ld(Qh + tn0 * 16 * HS_STRIDE + ks * 32, HS_STRIDE);
        v16h b1 = frag_ld(Qh + (tn0 + 1) * 16 * HS_STRIDE + ks * 32, HS_STRIDE);
        acc0 = wmma_f16(a, b0, acc0);
        acc1 = wmma_f16(a, b1, acc1);
      }
      #pragma unroll
      for (int u = 0; u < 2; ++u) {
        const v8f acc = u ? acc1 : acc0;
        const int j = (tn0 + u) * 16 + nl;
        #pragma unroll
        for (int r = 0; r < 8; ++r) {
          const int i = tm * 16 + r + 8 * half;
          const int ridx = ((i >> 3) - (j >> 3) + 7) * 15 + ((i & 7) - (j & 7) + 7);
          Sf[i * SF_STRIDE + j] = acc[r] + rel_table[ridx * 8 + h]
                                + maskw[i * 64 + j];
        }
      }
    }
    __syncthreads();

    // ---- row softmax (one thread per row; recompute exp to avoid spills)
    if (tid < 64) {
      const float* sr = Sf + tid * SF_STRIDE;
      float mx = sr[0];
      for (int j = 1; j < 64; ++j) mx = fmaxf(mx, sr[j]);
      float sum = 0.f;
      for (int j = 0; j < 64; ++j) sum += __expf(sr[j] - mx);
      const float inv = 1.f / sum;
      _Float16* sd = Sh + tid * HS_STRIDE;
      for (int j = 0; j < 64; ++j) sd[j] = (_Float16)(__expf(sr[j] - mx) * inv);
    }
    __syncthreads();

    // ---- Y^T[d][i] = sum_j Vt[d][j] * Sh[i][j]   (A=Vt, B^T=Sh, both row-major)
    //      C-tile consecutive-m direction == channel d -> packed b128 stores
    {
      const int tm = wave >> 1, tn0 = (wave & 1) * 2;   // tm: d-tile, tn: i-tile
      v8f acc0 = {}, acc1 = {};
      #pragma unroll
      for (int ks = 0; ks < 2; ++ks) {
        v16h a  = frag_ld(Vt + tm * 16 * HS_STRIDE + ks * 32, HS_STRIDE);
        v16h b0 = frag_ld(Sh + tn0 * 16 * HS_STRIDE + ks * 32, HS_STRIDE);
        v16h b1 = frag_ld(Sh + (tn0 + 1) * 16 * HS_STRIDE + ks * 32, HS_STRIDE);
        acc0 = wmma_f16(a, b0, acc0);
        acc1 = wmma_f16(a, b1, acc1);
      }
      #pragma unroll
      for (int u = 0; u < 2; ++u) {
        const v8f acc = u ? acc1 : acc0;
        const int i = (tn0 + u) * 16 + nl;
        v8h hv;
        #pragma unroll
        for (int r = 0; r < 8; ++r) hv[r] = (_Float16)acc[r];
        *(v8h*)(Ys + i * XS_STRIDE + h * 64 + tm * 16 + 8 * half) = hv;
      }
    }
    __syncthreads();   // protect Qh/Kh/Vt/Sf/Sh before next head
  }

  // ---- output projection (64x512x512) + window-decode / roll(+4,+4) scatter
  // Wave: tm = wave>>1; 4 groups x 4 n-tiles, A fragment shared across 4 tiles.
  {
    const int tm = wave >> 1;
    const _Float16* ab = Ys + tm * 16 * XS_STRIDE;
    for (int g = 0; g < 4; ++g) {
      const int tn0 = (wave & 1) * 16 + g * 4;
      v8f acc[4] = {{}, {}, {}, {}};
      const _Float16* wb = outw_h + (size_t)(tn0 * 16) * 512;
      #pragma unroll 2
      for (int ks = 0; ks < 16; ++ks) {
        v16h a = frag_ld(ab + ks * 32, XS_STRIDE);
        #pragma unroll
        for (int u = 0; u < 4; ++u) {
          v16h bm = frag_ld(wb + (size_t)u * 16 * 512 + ks * 32, 512);
          acc[u] = wmma_f16(a, bm, acc[u]);
        }
      }
      #pragma unroll
      for (int u = 0; u < 4; ++u) {
        const int o = (tn0 + u) * 16 + nl;
        const float bo = out_b[o];
        #pragma unroll
        for (int r = 0; r < 8; ++r) {
          const int tok = tm * 16 + r + 8 * half;
          const int tr = tok >> 3, tc = tok & 7;
          const int dr = (wr * 8 + tr + 4) & 63;
          const int dc = (wc * 8 + tc + 4) & 63;
          out[(((size_t)b * 4096) + dr * 64 + dc) * 512 + o] = acc[u][r] + bo;
        }
      }
    }
  }
}

extern "C" void kernel_launch(void* const* d_in, const int* in_sizes, int n_in,
                              void* d_out, int out_size, void* d_ws, size_t ws_size,
                              hipStream_t stream) {
  const float* x       = (const float*)d_in[0];
  const float* mask    = (const float*)d_in[1];
  const float* wqkv_w  = (const float*)d_in[2];
  const float* wqkv_b  = (const float*)d_in[3];
  const float* rel_tab = (const float*)d_in[4];
  const float* out_w   = (const float*)d_in[5];
  const float* out_b   = (const float*)d_in[6];
  float* out = (float*)d_out;

  _Float16* wq_h = (_Float16*)d_ws;                 // 3*512*512 f16 = 1.5 MB
  _Float16* ow_h = wq_h + 3 * 512 * 512;            // 512*512 f16 = 0.5 MB

  convert_weights<<<3072, 256, 0, stream>>>(wqkv_w, out_w, wq_h, ow_h);
  swin_attn_kernel<<<1024, 256, 0, stream>>>(x, mask, wq_h, wqkv_b, rel_tab,
                                             ow_h, out_b, out);
  (void)in_sizes; (void)n_in; (void)out_size; (void)ws_size;
}
